// MultiHeadAttention_58737972740646
// MI455X (gfx1250) — compile-verified, matrix operand order checked by reading, not run on hardware
//
#include <hip/hip_runtime.h>
#include <hip/hip_fp16.h>

typedef __attribute__((ext_vector_type(16))) _Float16 v16h;
typedef __attribute__((ext_vector_type(8)))  float    v8f;

#define D_MODEL 1024
#define SEQ     2048
#define NHEAD   16
#define DKH     64
#define BATCH   2
#define MROWS   (BATCH*SEQ)     /* 4096 */

/* ---------------- fragment helpers ----------------
 * 16-bit A/B fragment layout (CDNA5 ISA 7.12.2):
 *   lane = (row 0..15) + 16*kh ; elements 0..7 -> k = kh*8 + 0..7
 *                                elements 8..15 -> k = 16 + kh*8 + 0..7
 * so a fragment = two contiguous 16B runs at +kh*8 and +16+kh*8 halves. */
static __device__ __forceinline__ v16h ld_frag(const _Float16* p) {
  v16h r;
  ((uint4*)&r)[0] = *(const uint4*)p;
  ((uint4*)&r)[1] = *(const uint4*)(p + 16);
  return r;
}

/* ---------------- fp32 -> f16 conversion ---------------- */
__global__ __launch_bounds__(256) void cvt_f32_f16(const float* __restrict__ in,
                                                   _Float16* __restrict__ out, int n) {
  int i = (blockIdx.x * 256 + threadIdx.x) * 4;
  if (i + 3 < n) {
    float4 v = *(const float4*)(in + i);
    out[i + 0] = (_Float16)v.x; out[i + 1] = (_Float16)v.y;
    out[i + 2] = (_Float16)v.z; out[i + 3] = (_Float16)v.w;
  }
}

/* ---------------- WMMA GEMM: C = X @ W^T + bias ----------------
 * X: [4096 x 1024] (fp32 or f16), W: [1024 x 1024] f16 (row n = output col),
 * SPLIT_OUT: write f16 head-split [B,H,S,64]; else fp32 row-major [4096x1024]. */
#define BM 128
#define BN 128
#define BK 32
#define LDT 40   /* LDS row stride in halves: 32 + 8 pad (80B, 16B-aligned) */

template <bool X_F16, bool SPLIT_OUT>
__global__ __launch_bounds__(256)
void wmma_gemm_xwt(const void* __restrict__ Xv, const _Float16* __restrict__ W,
                   const float* __restrict__ bias, void* __restrict__ Outv) {
  __shared__ __align__(16) _Float16 As[BM * LDT];
  __shared__ __align__(16) _Float16 Bs[BN * LDT];

  const int tid  = threadIdx.x;
  const int wave = tid >> 5, lane = tid & 31;
  const int l16  = lane & 15, kh = lane >> 4;
  const int waveM = wave & 3;   /* 4 waves x 32 rows  */
  const int waveN = wave >> 2;  /* 2 waves x 64 cols  */
  const int m0 = blockIdx.y * BM;
  const int n0 = blockIdx.x * BN;

  v8f acc[2][4];
#pragma unroll
  for (int i = 0; i < 2; i++)
#pragma unroll
    for (int j = 0; j < 4; j++) acc[i][j] = (v8f){};

  for (int k0 = 0; k0 < D_MODEL; k0 += BK) {
    /* stage X tile (convert fp32->f16 if needed) */
    if constexpr (X_F16) {
      const _Float16* X = (const _Float16*)Xv;
#pragma unroll
      for (int i = 0; i < 2; i++) {
        int c = tid + i * 256, r = c >> 2, cc = c & 3;
        *(uint4*)(As + r * LDT + cc * 8) =
            *(const uint4*)(X + (size_t)(m0 + r) * D_MODEL + k0 + cc * 8);
      }
    } else {
      const float* X = (const float*)Xv;
#pragma unroll
      for (int i = 0; i < 4; i++) {
        int c = tid + i * 256, r = c >> 3, cc = c & 7;
        float4 d = *(const float4*)(X + (size_t)(m0 + r) * D_MODEL + k0 + cc * 4);
        _Float16* dst = As + r * LDT + cc * 4;
        dst[0] = (_Float16)d.x; dst[1] = (_Float16)d.y;
        dst[2] = (_Float16)d.z; dst[3] = (_Float16)d.w;
      }
    }
    /* stage W tile (rows n0..n0+127) */
#pragma unroll
    for (int i = 0; i < 2; i++) {
      int c = tid + i * 256, r = c >> 2, cc = c & 3;
      *(uint4*)(Bs + r * LDT + cc * 8) =
          *(const uint4*)(W + (size_t)(n0 + r) * D_MODEL + k0 + cc * 8);
    }
    /* prefetch next K-slice */
    if (k0 + BK < D_MODEL) {
      __builtin_prefetch(W + (size_t)(n0 + (tid >> 1)) * D_MODEL + k0 + BK, 0, 1);
    }
    __syncthreads();

    v16h a[2], b[4];
#pragma unroll
    for (int fi = 0; fi < 2; fi++)
      a[fi] = ld_frag(As + (waveM * 32 + fi * 16 + l16) * LDT + kh * 8);
#pragma unroll
    for (int fj = 0; fj < 4; fj++)
      b[fj] = ld_frag(Bs + (waveN * 64 + fj * 16 + l16) * LDT + kh * 8);
#pragma unroll
    for (int fi = 0; fi < 2; fi++)
#pragma unroll
      for (int fj = 0; fj < 4; fj++)
        acc[fi][fj] = __builtin_amdgcn_wmma_f32_16x16x32_f16(
            false, a[fi], false, b[fj], (short)0, acc[fi][fj], false, false);
    __syncthreads();
  }

  /* epilogue: C layout -> lane holds col n=l16, VGPR r holds row r + 8*kh */
#pragma unroll
  for (int fi = 0; fi < 2; fi++)
#pragma unroll
    for (int fj = 0; fj < 4; fj++) {
      int n = n0 + waveN * 64 + fj * 16 + l16;
      float bv = bias[n];
#pragma unroll
      for (int r = 0; r < 8; r++) {
        int m = m0 + waveM * 32 + fi * 16 + r + kh * 8;
        float val = acc[fi][fj][r] + bv;
        if constexpr (SPLIT_OUT) {
          _Float16* O = (_Float16*)Outv;
          int bb = m >> 11, s = m & (SEQ - 1);
          int h = n >> 6, dk = n & (DKH - 1);
          O[(((size_t)bb * NHEAD + h) * SEQ + s) * DKH + dk] = (_Float16)val;
        } else {
          ((float*)Outv)[(size_t)m * D_MODEL + n] = val;
        }
      }
    }
}

/* ---------------- flash attention ----------------
 * grid (SEQ/64, B*H), 128 threads (4 waves x 16 Q rows). Online softmax,
 * scores never touch HBM. Q,K,V: f16 head-split [bh][s][64]. */
#define LDV 72
#define LDP 72

__global__ __launch_bounds__(128)
void flash_attn(const _Float16* __restrict__ Q, const _Float16* __restrict__ K,
                const _Float16* __restrict__ V, _Float16* __restrict__ Out) {
  __shared__ __align__(16) _Float16 Vt[64 * LDV];      /* V tile transposed */
  __shared__ __align__(16) _Float16 Pw[4][16 * LDP];   /* per-wave P tile   */

  const int tid  = threadIdx.x;
  const int wave = tid >> 5, lane = tid & 31;
  const int l16  = lane & 15, kh = lane >> 4;
  const int bh = blockIdx.y, b = bh >> 4, h = bh & 15;
  const int qBase = blockIdx.x * 64 + wave * 16;
  const size_t headOff = (size_t)bh * SEQ * DKH;

  /* Q fragments: 16 rows x 64 d = two 16x32 A-frags, kept in VGPRs */
  v16h qf[2];
  {
    const _Float16* qp = Q + headOff + (size_t)(qBase + l16) * DKH;
#pragma unroll
    for (int ks = 0; ks < 2; ks++) qf[ks] = ld_frag(qp + ks * 32 + kh * 8);
  }

  v8f o[4];
#pragma unroll
  for (int i = 0; i < 4; i++) o[i] = (v8f){};
  float mrun[8], lrun[8];
#pragma unroll
  for (int r = 0; r < 8; r++) { mrun[r] = -1.0e30f; lrun[r] = 0.0f; }

  for (int j = 0; j < SEQ; j += 64) {
    /* stage V tile transposed: Vt[d][n] */
#pragma unroll
    for (int i = 0; i < 4; i++) {
      int c = tid + i * 128, n = c >> 3, dc = c & 7;
      uint4 d4 = *(const uint4*)(V + headOff + (size_t)(j + n) * DKH + dc * 8);
      _Float16 tmp[8]; *(uint4*)tmp = d4;
#pragma unroll
      for (int e = 0; e < 8; e++) Vt[(dc * 8 + e) * LDV + n] = tmp[e];
    }
    __syncthreads();

    /* S = Q K^T : K rows are B-frag columns (contiguous from global) */
    v8f s[4];
#pragma unroll
    for (int nf = 0; nf < 4; nf++) s[nf] = (v8f){};
#pragma unroll
    for (int ks = 0; ks < 2; ks++)
#pragma unroll
      for (int nf = 0; nf < 4; nf++) {
        v16h kf = ld_frag(K + headOff + (size_t)(j + nf * 16 + l16) * DKH +
                          ks * 32 + kh * 8);
        s[nf] = __builtin_amdgcn_wmma_f32_16x16x32_f16(
            false, qf[ks], false, kf, (short)0, s[nf], false, false);
      }

    /* online softmax (scale 1/sqrt(64) = 0.125) */
#pragma unroll
    for (int nf = 0; nf < 4; nf++)
#pragma unroll
      for (int r = 0; r < 8; r++) s[nf][r] *= 0.125f;

    float newm[8], alpha[8], rs[8];
#pragma unroll
    for (int r = 0; r < 8; r++) {
      float mx = fmaxf(fmaxf(s[0][r], s[1][r]), fmaxf(s[2][r], s[3][r]));
#pragma unroll
      for (int off = 8; off >= 1; off >>= 1)
        mx = fmaxf(mx, __shfl_xor(mx, off, 32));
      newm[r]  = fmaxf(mrun[r], mx);
      alpha[r] = __expf(mrun[r] - newm[r]);
      mrun[r]  = newm[r];
      rs[r]    = 0.0f;
    }
#pragma unroll
    for (int nf = 0; nf < 4; nf++)
#pragma unroll
      for (int r = 0; r < 8; r++) {
        float p = __expf(s[nf][r] - newm[r]);
        rs[r] += p;
        Pw[wave][(r + kh * 8) * LDP + nf * 16 + l16] = (_Float16)p;
      }
#pragma unroll
    for (int r = 0; r < 8; r++) {
#pragma unroll
      for (int off = 8; off >= 1; off >>= 1) rs[r] += __shfl_xor(rs[r], off, 32);
      lrun[r] = lrun[r] * alpha[r] + rs[r];
    }
#pragma unroll
    for (int df = 0; df < 4; df++)
#pragma unroll
      for (int r = 0; r < 8; r++) o[df][r] *= alpha[r];

    /* intra-wave LDS round trip: ensure P stores land before frag reads */
    asm volatile("s_wait_dscnt 0" ::: "memory");

    /* O += P V : A-frag from per-wave P, B-frag from transposed V */
#pragma unroll
    for (int ks = 0; ks < 2; ks++) {
      v16h pa = ld_frag(&Pw[wave][l16 * LDP + ks * 32 + kh * 8]);
#pragma unroll
      for (int df = 0; df < 4; df++) {
        v16h vb = ld_frag(&Vt[(df * 16 + l16) * LDV + ks * 32 + kh * 8]);
        o[df] = __builtin_amdgcn_wmma_f32_16x16x32_f16(
            false, pa, false, vb, (short)0, o[df], false, false);
      }
    }
    __syncthreads();   /* protect Vt before next tile overwrites it */
  }

  /* normalize and write merged-head f16 [b][s][h*64+d] */
#pragma unroll
  for (int df = 0; df < 4; df++)
#pragma unroll
    for (int r = 0; r < 8; r++) {
      int q  = qBase + r + kh * 8;
      int col = h * DKH + df * 16 + l16;
      float inv = 1.0f / (lrun[r] + 1e-9f);
      Out[((size_t)b * SEQ + q) * D_MODEL + col] = (_Float16)(o[df][r] * inv);
    }
}

/* ---------------- host launcher ---------------- */
extern "C" void kernel_launch(void* const* d_in, const int* in_sizes, int n_in,
                              void* d_out, int out_size, void* d_ws, size_t ws_size,
                              hipStream_t stream) {
  const float* q  = (const float*)d_in[0];
  const float* k  = (const float*)d_in[1];
  const float* v  = (const float*)d_in[2];
  const float* Wq = (const float*)d_in[3];
  const float* bq = (const float*)d_in[4];
  const float* Wk = (const float*)d_in[5];
  const float* bk = (const float*)d_in[6];
  const float* Wv = (const float*)d_in[7];
  const float* bv = (const float*)d_in[8];
  const float* Wo = (const float*)d_in[9];
  const float* bo = (const float*)d_in[10];

  const size_t WN = (size_t)D_MODEL * D_MODEL;  /* 1M elems  */
  const size_t TN = (size_t)MROWS * D_MODEL;    /* 4M elems  */

  char* ws = (char*)d_ws;
  _Float16* Wq16 = (_Float16*)(ws + 0 * WN * 2);
  _Float16* Wk16 = (_Float16*)(ws + 1 * WN * 2);
  _Float16* Wv16 = (_Float16*)(ws + 2 * WN * 2);
  _Float16* Wo16 = (_Float16*)(ws + 3 * WN * 2);
  char* p = ws + 4 * WN * 2;
  _Float16* Q16 = (_Float16*)(p + 0 * TN * 2);
  _Float16* K16 = (_Float16*)(p + 1 * TN * 2);
  _Float16* V16 = (_Float16*)(p + 2 * TN * 2);
  _Float16* A16 = (_Float16*)(p + 3 * TN * 2);

  /* weights fp32 -> f16 */
  int gcv = (int)(WN / 1024);
  cvt_f32_f16<<<gcv, 256, 0, stream>>>(Wq, Wq16, (int)WN);
  cvt_f32_f16<<<gcv, 256, 0, stream>>>(Wk, Wk16, (int)WN);
  cvt_f32_f16<<<gcv, 256, 0, stream>>>(Wv, Wv16, (int)WN);
  cvt_f32_f16<<<gcv, 256, 0, stream>>>(Wo, Wo16, (int)WN);

  /* QKV projections -> head-split f16 */
  dim3 gB(D_MODEL / BN, MROWS / BM);  /* (8, 32) */
  wmma_gemm_xwt<false, true><<<gB, 256, 0, stream>>>(q, Wq16, bq, Q16);
  wmma_gemm_xwt<false, true><<<gB, 256, 0, stream>>>(k, Wk16, bk, K16);
  wmma_gemm_xwt<false, true><<<gB, 256, 0, stream>>>(v, Wv16, bv, V16);

  /* flash attention */
  flash_attn<<<dim3(SEQ / 64, BATCH * NHEAD), 128, 0, stream>>>(Q16, K16, V16, A16);

  /* output projection -> fp32 d_out */
  wmma_gemm_xwt<true, false><<<gB, 256, 0, stream>>>(A16, Wo16, bo, d_out);
}
